// MultiHeadAttentionShuffle_27324581937448
// MI455X (gfx1250) — matrix operand order repaired and checked by reading, not compiled, not run on hardware
//
#include <hip/hip_runtime.h>
#include <hip/hip_bf16.h>

// Shapes (fixed by the reference): BS=4, SL=128, D_IN=D_MODEL=512, H=8, DP=64
#define BS_ 4
#define SL_ 128
#define DM_ 512
#define H_  8
#define DP_ 64

typedef __bf16        bf16x16 __attribute__((ext_vector_type(16)));
typedef float         f32x8   __attribute__((ext_vector_type(8)));
typedef unsigned int  u32x4   __attribute__((ext_vector_type(4)));

union Frag {            // 16 bf16 = one WMMA A/B operand (8 VGPRs)
  bf16x16 v;
  u32x4   q[2];
};

__device__ __forceinline__ unsigned short f2bf(float f) {   // RNE fp32 -> bf16
  unsigned u = __float_as_uint(f);
  u += 0x7FFFu + ((u >> 16) & 1u);
  return (unsigned short)(u >> 16);
}

// ---------------- threefry2x32 (jax PRNG core) -------------------------------
__device__ __forceinline__ void threefry2x32(unsigned k0, unsigned k1,
                                             unsigned x0, unsigned x1,
                                             unsigned& o0, unsigned& o1) {
  unsigned ks[3] = {k0, k1, k0 ^ k1 ^ 0x1BD11BDAu};
  const unsigned rotA[4] = {13, 15, 26, 6};
  const unsigned rotB[4] = {17, 29, 16, 24};
  x0 += ks[0]; x1 += ks[1];
#pragma unroll
  for (int g = 0; g < 5; ++g) {
    const unsigned* rr = (g & 1) ? rotB : rotA;
#pragma unroll
    for (int i = 0; i < 4; ++i) {
      x0 += x1;
      x1 = (x1 << rr[i]) | (x1 >> (32 - rr[i]));
      x1 ^= x0;
    }
    x0 += ks[(g + 1) % 3];
    x1 += ks[(g + 2) % 3] + (unsigned)(g + 1);
  }
  o0 = x0; o1 = x1;
}

// Recompute idx_q/idx_k/idx_v = argsort(where(j<t, U(key42,i)[t,j], j+2)) rows.
// fold_in(key(42), i) -> threefry((0,42),(0,i)); uniform bits per element e:
// partitionable counter mode: enc(hi=0, lo=e), bits = o0 ^ o1.
__global__ void k_idx(int* __restrict__ idx) {
  const int t = blockIdx.x, which = blockIdx.y, j = threadIdx.x;
  unsigned fk0, fk1;
  threefry2x32(0u, 42u, 0u, (unsigned)which, fk0, fk1);
  unsigned e = (unsigned)(t * SL_ + j);
  unsigned o0, o1;
  threefry2x32(fk0, fk1, 0u, e, o0, o1);
  unsigned bits = o0 ^ o1;
  float u = __uint_as_float((bits >> 9) | 0x3F800000u) - 1.0f;
  float key = (j < t) ? u : ((float)j + 2.0f);

  __shared__ float kv[SL_];
  __shared__ int   so[SL_];
  kv[j] = key;
  __syncthreads();
  int rank = 0;
  for (int p = 0; p < SL_; ++p) {                 // stable rank
    float o = kv[p];
    rank += (o < key) || (o == key && p < j);
  }
  so[rank] = j;
  __syncthreads();
  idx[(which * SL_ + t) * SL_ + j] = so[j];
}

// ---------------- fp32 -> bf16 converts --------------------------------------
__global__ void k_cvt(const float* __restrict__ in, unsigned short* __restrict__ out, int n) {
  int i = blockIdx.x * blockDim.x + threadIdx.x;
  if (i < n) out[i] = f2bf(in[i]);
}
// W is (K rows, N cols) row-major; emit WT (N, K) bf16 so B-operand loads are contiguous
__global__ void k_cvtT(const float* __restrict__ in, unsigned short* __restrict__ out, int K, int N) {
  int i = blockIdx.x * blockDim.x + threadIdx.x;
  if (i < K * N) {
    int kk = i % K, n = i / K;
    out[i] = f2bf(in[kk * N + n]);
  }
}

// ---------------- generic bf16 WMMA GEMM: C = A(MxK) * BT(NxK)^T + bias ------
// One wave per 16x16 tile of C. fp32 accumulate; optional fp32 and bf16 outputs.
__global__ void __launch_bounds__(32)
k_gemm(const unsigned short* __restrict__ A, const unsigned short* __restrict__ BT,
       const float* __restrict__ bias, float* __restrict__ Cf,
       unsigned short* __restrict__ Cb, int M, int N, int K) {
  const int L = threadIdx.x, lo = L & 15, hi = L >> 4;
  const int n0 = blockIdx.x * 16, m0 = blockIdx.y * 16;
  const unsigned short* arow = A  + (size_t)(m0 + lo) * K;   // A: lane = row M
  const unsigned short* brow = BT + (size_t)(n0 + lo) * K;   // B: lane = col N
  f32x8 acc = {};
  for (int k0 = 0; k0 < K; k0 += 32) {
    Frag a, b;
    // A 16x32 bf16 layout: lane elems i -> K = k0 + hi*16 + i (contiguous)
    a.q[0] = *(const u32x4*)(arow + k0 + hi * 16);
    a.q[1] = *(const u32x4*)(arow + k0 + hi * 16 + 8);
    // B 32x16 bf16 layout: lane elems i -> K = k0 + hi*16 + i (contiguous)
    b.q[0] = *(const u32x4*)(brow + k0 + hi * 16);
    b.q[1] = *(const u32x4*)(brow + k0 + hi * 16 + 8);
    acc = __builtin_amdgcn_wmma_f32_16x16x32_bf16(false, a.v, false, b.v,
                                                  (short)0, acc, false, false);
  }
  float bb = bias ? bias[n0 + lo] : 0.0f;
#pragma unroll
  for (int r = 0; r < 8; ++r) {                  // C: VGPR r -> row r + hi*8, col = lo
    size_t o = (size_t)(m0 + r + hi * 8) * N + (n0 + lo);
    float val = acc[r] + bb;
    if (Cf) Cf[o] = val;
    if (Cb) Cb[o] = f2bf(val);
  }
}

// ---------------- attention: logits + mask + softmax + full attn output ------
// Block = 8 waves, one block per (b,t,h). Wave w owns q rows [16w,16w+16).
__global__ void __launch_bounds__(256)
k_attn(const unsigned short* __restrict__ qpb, const unsigned short* __restrict__ kpb,
       const int* __restrict__ idx, float* __restrict__ attn) {
  const int h = blockIdx.x, t = blockIdx.y, b = blockIdx.z;
  const int w = threadIdx.x >> 5;
  const int L = threadIdx.x & 31, lo = L & 15, hi = L >> 4;
  const int* idxq = idx + (0 * SL_ + t) * SL_;
  const int* idxk = idx + (1 * SL_ + t) * SL_;

  // Gather this wave's Q rows (64 bf16 per row) through idx_q; zero rows s>t.
  Frag a0, a1;
  {
    int srow = w * 16 + lo;
    if (srow <= t) {
      const unsigned short* qr = qpb + (size_t)(b * SL_ + idxq[srow]) * DM_ + h * DP_;
      a0.q[0] = *(const u32x4*)(qr + hi * 16);
      a0.q[1] = *(const u32x4*)(qr + hi * 16 + 8);
      a1.q[0] = *(const u32x4*)(qr + 32 + hi * 16);
      a1.q[1] = *(const u32x4*)(qr + 32 + hi * 16 + 8);
    } else {
      a0.q[0] = (u32x4)0u; a0.q[1] = (u32x4)0u;
      a1.q[0] = (u32x4)0u; a1.q[1] = (u32x4)0u;
    }
  }

  float lg[8][8];                                // [k-tile][C row-vgpr]
#pragma unroll
  for (int kt = 0; kt < 8; ++kt) {
    int col = kt * 16 + lo;
    int krow = (col <= t) ? idxk[col] : col;     // cols > t are masked anyway
    const unsigned short* kr = kpb + (size_t)(b * SL_ + krow) * DM_ + h * DP_;
    Frag b0, b1;                                 // B = Kh^T (d x k_s): K-dim contiguous in kp row
    b0.q[0] = *(const u32x4*)(kr + hi * 16);
    b0.q[1] = *(const u32x4*)(kr + hi * 16 + 8);
    b1.q[0] = *(const u32x4*)(kr + 32 + hi * 16);
    b1.q[1] = *(const u32x4*)(kr + 32 + hi * 16 + 8);
    f32x8 c = {};
    c = __builtin_amdgcn_wmma_f32_16x16x32_bf16(false, a0.v, false, b0.v, (short)0, c, false, false);
    c = __builtin_amdgcn_wmma_f32_16x16x32_bf16(false, a1.v, false, b1.v, (short)0, c, false, false);
    float mterm = (col > t) ? -1.0e9f : 0.0f;    // causal mask
#pragma unroll
    for (int r = 0; r < 8; ++r) lg[kt][r] = c[r] * 0.125f + mterm;   // /sqrt(64)
  }

  // Row softmax: row m = 16w + r + hi*8 is held by the 16 lanes sharing `hi`.
  const size_t base = ((((size_t)b * SL_ + t) * H_ + h) * SL_) * SL_;
#pragma unroll
  for (int r = 0; r < 8; ++r) {
    float mx = lg[0][r];
#pragma unroll
    for (int kt = 1; kt < 8; ++kt) mx = fmaxf(mx, lg[kt][r]);
#pragma unroll
    for (int off = 1; off <= 8; off <<= 1) mx = fmaxf(mx, __shfl_xor(mx, off, 32));
    float s = 0.0f;
#pragma unroll
    for (int kt = 0; kt < 8; ++kt) { float e = expf(lg[kt][r] - mx); lg[kt][r] = e; s += e; }
#pragma unroll
    for (int off = 1; off <= 8; off <<= 1) s += __shfl_xor(s, off, 32);
    float inv = 1.0f / s;
    float* orow = attn + base + (size_t)(w * 16 + r + hi * 8) * SL_;
#pragma unroll
    for (int kt = 0; kt < 8; ++kt) orow[kt * 16 + lo] = lg[kt][r] * inv;
  }
}

// ---------------- diagonal context: eff[b,t,:] = attn[b,t,h,t,:] @ Vh --------
__global__ void __launch_bounds__(512)
k_eff(const float* __restrict__ attn, const float* __restrict__ vpf,
      const int* __restrict__ idx, unsigned short* __restrict__ effb) {
  const int bt = blockIdx.x, b = bt >> 7, t = bt & 127;
  const int c = threadIdx.x, h = c >> 6;         // c = h*64 + d
  const int* idxv = idx + (2 * SL_ + t) * SL_;
  __shared__ int siv[SL_];
  if (c < SL_) siv[c] = idxv[c];
  __syncthreads();
  const float* arow = attn + ((((size_t)b * SL_ + t) * H_ + h) * SL_ + t) * SL_;
  float s = 0.0f;
  for (int k = 0; k < SL_; ++k)                  // attn is exactly 0 beyond t
    s = fmaf(arow[k], vpf[(size_t)(b * SL_ + siv[k]) * DM_ + c], s);
  effb[(size_t)bt * DM_ + c] = f2bf(s);
}

// -----------------------------------------------------------------------------
extern "C" void kernel_launch(void* const* d_in, const int* in_sizes, int n_in,
                              void* d_out, int out_size, void* d_ws, size_t ws_size,
                              hipStream_t stream) {
  (void)in_sizes; (void)n_in; (void)out_size; (void)ws_size;
  const float* q    = (const float*)d_in[0];
  const float* k    = (const float*)d_in[1];
  const float* v    = (const float*)d_in[2];
  /* d_in[3] = mask (recomputed on device) */
  const float* wq_w = (const float*)d_in[4];
  const float* wq_b = (const float*)d_in[5];
  const float* wk_w = (const float*)d_in[6];
  const float* wk_b = (const float*)d_in[7];
  const float* wv_w = (const float*)d_in[8];
  const float* wv_b = (const float*)d_in[9];
  const float* dw   = (const float*)d_in[10];
  const float* db   = (const float*)d_in[11];

  float* out  = (float*)d_out;                    // (4,128,512)
  float* attn = out + BS_ * SL_ * DM_;            // (4,128,8,128,128)

  const int NE = DM_ * DM_;                       // 262144 (= BS*SL*DM too)
  char* ws = (char*)d_ws;                         // all offsets 16B-aligned
  int*            idx  = (int*)(ws + 0);                       // 3*128*128 i32
  unsigned short* qb   = (unsigned short*)(ws + 196608);
  unsigned short* kb   = (unsigned short*)(ws + 720896);
  unsigned short* vb   = (unsigned short*)(ws + 1245184);
  unsigned short* wqT  = (unsigned short*)(ws + 1769472);
  unsigned short* wkT  = (unsigned short*)(ws + 2293760);
  unsigned short* wvT  = (unsigned short*)(ws + 2818048);
  unsigned short* dT   = (unsigned short*)(ws + 3342336);
  unsigned short* qpb  = (unsigned short*)(ws + 3866624);
  unsigned short* kpb  = (unsigned short*)(ws + 4390912);
  float*          vpf  = (float*)(ws + 4915200);
  unsigned short* effb = (unsigned short*)(ws + 5963776);      // end: 6488064 B

  // 1) jax-threefry shuffle permutations
  k_idx<<<dim3(SL_, 3), SL_, 0, stream>>>(idx);

  // 2) bf16 copies (activations row-major; weights transposed)
  const int cb = 256, cg = (NE + cb - 1) / cb;
  k_cvt <<<cg, cb, 0, stream>>>(q, qb, NE);
  k_cvt <<<cg, cb, 0, stream>>>(k, kb, NE);
  k_cvt <<<cg, cb, 0, stream>>>(v, vb, NE);
  k_cvtT<<<cg, cb, 0, stream>>>(wq_w, wqT, DM_, DM_);
  k_cvtT<<<cg, cb, 0, stream>>>(wk_w, wkT, DM_, DM_);
  k_cvtT<<<cg, cb, 0, stream>>>(wv_w, wvT, DM_, DM_);
  k_cvtT<<<cg, cb, 0, stream>>>(dw,   dT,  DM_, DM_);

  // 3) projections (WMMA): qp/kp kept bf16 for attention, vp kept fp32 for ctx
  dim3 gg(DM_ / 16, DM_ / 16);
  k_gemm<<<gg, 32, 0, stream>>>(qb, wqT, wq_b, nullptr, qpb, BS_ * SL_, DM_, DM_);
  k_gemm<<<gg, 32, 0, stream>>>(kb, wkT, wk_b, nullptr, kpb, BS_ * SL_, DM_, DM_);
  k_gemm<<<gg, 32, 0, stream>>>(vb, wvT, wv_b, vpf,     nullptr, BS_ * SL_, DM_, DM_);

  // 4) full attention tensor (bandwidth-bound: 268 MB of output)
  k_attn<<<dim3(H_, SL_, BS_), 256, 0, stream>>>(qpb, kpb, idx, attn);

  // 5) diagonal context, then dense output projection (WMMA)
  k_eff<<<BS_ * SL_, DM_, 0, stream>>>(attn, vpf, idx, effb);
  k_gemm<<<gg, 32, 0, stream>>>(effb, dT, db, out, nullptr, BS_ * SL_, DM_, DM_);
}
